// MultiheadAttention_67783173865769
// MI455X (gfx1250) — matrix operand order repaired and checked by reading, not compiled, not run on hardware
//
#include <hip/hip_runtime.h>
#include <hip/hip_bf16.h>

typedef __bf16 bf16;
typedef __attribute__((ext_vector_type(16))) __bf16 v16bf;
typedef __attribute__((ext_vector_type(8)))  __bf16 v8bf;
typedef __attribute__((ext_vector_type(4)))  __bf16 v4bf;
typedef __attribute__((ext_vector_type(8)))  float  v8f;

typedef __attribute__((ext_vector_type(4))) unsigned int u32x4;
typedef __attribute__((ext_vector_type(8))) int          i32x8;
typedef __attribute__((ext_vector_type(4))) int          i32x4;

#ifndef __has_builtin
#define __has_builtin(x) 0
#endif
#if __has_builtin(__builtin_amdgcn_tensor_load_to_lds) && \
    __has_builtin(__builtin_amdgcn_s_wait_tensorcnt)
#define HAVE_TDM 1
#else
#define HAVE_TDM 0
#endif

#define LDSTR 72   // LDS row stride (elements); 72*2B = 144B keeps 16B alignment

static __device__ __forceinline__ bf16 f2bf(float x) { return (bf16)x; }

// A-fragment (16x32 bf16, MxK) from LDS, base points at [mbase*stride + kbase]
// ISA 7.12.2: lanes 0-15 hold M=lane, K {0..7, 16..23}; lanes 16-31 K {8..15, 24..31}
static __device__ __forceinline__ v16bf lds_load_a(const bf16* base, int stride, int lane) {
  int r  = lane & 15;
  int kb = (lane >> 4) << 3;        // 0 or 8
  const bf16* p = base + r * stride + kb;
  union { v16bf v; v8bf h[2]; } u;
  u.h[0] = *(const v8bf*)(p);
  u.h[1] = *(const v8bf*)(p + 16);
  return u.v;
}

// B-fragment (32x16 bf16, KxN) from LDS holding B^T (N rows of K elements)
// lanes 0-15: N=lane, K {0..7, 16..23} ; lanes 16-31: N=lane-16, K {8..15, 24..31}
static __device__ __forceinline__ v16bf lds_load_b(const bf16* baseT, int stride, int lane) {
  int n  = lane & 15;
  int kb = (lane >> 4) << 3;        // 0 or 8
  const bf16* p = baseT + n * stride + kb;
  union { v16bf v; v8bf h[2]; } u;
  u.h[0] = *(const v8bf*)(p);
  u.h[1] = *(const v8bf*)(p + 16);
  return u.v;
}

#define WMMA_BF16(a, b, c) \
  __builtin_amdgcn_wmma_f32_16x16x32_bf16(false, (a), false, (b), (short)0, (c), false, false)

#if HAVE_TDM
// TDM 2-D bf16 tile load: global (tile_d1 rows x tile_d0 elems, row stride
// stride_elems) -> LDS at lds_off with +16B row padding (pitch = LDSTR*2B).
// D# layout per CDNA5 ISA ch.8: group0 {flags, lds_addr, gaddr_lo, gaddr_hi|type},
// group1 {w0 flags/data_size/pad, dims/tiles/strides}.
static __device__ __forceinline__ void tdm_load_tile_bf16(
    const bf16* gptr, unsigned lds_off,
    unsigned tile_d0, unsigned tile_d1, unsigned long long stride_elems)
{
  unsigned long long ga = (unsigned long long)(size_t)gptr;
  u32x4 g0;
  g0[0] = 1u;                                           // count=1 (valid user D#)
  g0[1] = lds_off;                                      // LDS byte address
  g0[2] = (unsigned)ga;                                 // global addr [31:0]
  g0[3] = (unsigned)((ga >> 32) & 0x1FFFFFFu)           // global addr [56:32]
        | 0x80000000u;                                  // type=2 ("image")
  i32x8 g1;
  unsigned w0 = (1u << 16)                              // data_size=1 (2 bytes)
              | (1u << 20)                              // pad_enable
              | (4u << 22)                              // pad_interval: 32 DWORDs
              | (3u << 25);                             // pad_amount: 4 DWORDs (16B)
  g1[0] = (int)w0;
  g1[1] = (int)((tile_d0 & 0xFFFFu) << 16);             // tensor_dim0[15:0] (=tile)
  g1[2] = (int)(((tile_d0 >> 16) & 0xFFFFu) | ((tile_d1 & 0xFFFFu) << 16)); // dim0 hi / dim1 lo
  g1[3] = (int)(((tile_d1 >> 16) & 0xFFFFu) | ((tile_d0 & 0xFFFFu) << 16)); // dim1 hi / tile_dim0
  g1[4] = (int)(tile_d1 & 0xFFFFu);                     // tile_dim1; tile_dim2=0
  g1[5] = (int)(unsigned)(stride_elems & 0xFFFFFFFFu);  // tensor_dim0_stride lo32
  g1[6] = (int)(unsigned)((stride_elems >> 32) & 0xFFFFu); // stride hi16; dim1_stride lo=0
  g1[7] = 0;
  i32x4 z4 = {0, 0, 0, 0};
#if __clang_major__ >= 23
  i32x8 z8 = {0, 0, 0, 0, 0, 0, 0, 0};
  __builtin_amdgcn_tensor_load_to_lds(g0, g1, z4, z4, z8, 0);
#else
  __builtin_amdgcn_tensor_load_to_lds(g0, g1, z4, z4, 0);
#endif
}
#endif

// ---------------------------------------------------------------------------
// Kernel 0: one-time weight transpose + f32->bf16 convert.
// W (1024x1024, KxN row-major f32) -> WT (1024x1024, NxK row-major bf16).
// ---------------------------------------------------------------------------
__global__ __launch_bounds__(256)
void mha_transpose_w(const float* __restrict__ W0, const float* __restrict__ W1,
                     const float* __restrict__ W2, const float* __restrict__ W3,
                     bf16* __restrict__ T0, bf16* __restrict__ T1,
                     bf16* __restrict__ T2, bf16* __restrict__ T3)
{
  const int z = blockIdx.z;
  const float* W = (z == 0) ? W0 : (z == 1) ? W1 : (z == 2) ? W2 : W3;
  bf16*       WT = (z == 0) ? T0 : (z == 1) ? T1 : (z == 2) ? T2 : T3;

  __shared__ float sT[64][65];
  const int tid = threadIdx.x;
  const int n0 = blockIdx.x * 64, k0 = blockIdx.y * 64;

  float4 r[4];
  #pragma unroll
  for (int i = 0; i < 4; ++i) {             // coalesced f32 reads along N
    int idx = tid + i * 256;                // 1024 float4 slots
    int k = idx >> 4, c4 = idx & 15;
    r[i] = *(const float4*)(W + (size_t)(k0 + k) * 1024 + n0 + c4 * 4);
  }
  #pragma unroll
  for (int i = 0; i < 4; ++i) {
    int idx = tid + i * 256;
    int k = idx >> 4, c4 = idx & 15;
    sT[k][c4 * 4 + 0] = r[i].x; sT[k][c4 * 4 + 1] = r[i].y;
    sT[k][c4 * 4 + 2] = r[i].z; sT[k][c4 * 4 + 3] = r[i].w;
  }
  __syncthreads();
  #pragma unroll
  for (int i = 0; i < 2; ++i) {             // coalesced bf16 writes along K
    int idx = tid + i * 256;                // 512 chunks of 8
    int n = idx >> 3, c8 = idx & 7;
    v8bf o;
    #pragma unroll
    for (int j = 0; j < 8; ++j) o[j] = f2bf(sT[c8 * 8 + j][n]);
    *(v8bf*)(WT + (size_t)(n0 + n) * 1024 + k0 + c8 * 8) = o;
  }
}

// ---------------------------------------------------------------------------
// Kernel 1: fused Q/K/V projection, double-buffered LDS, tile M64 x N128.
// Each wave computes 32x64 (2 A-frags x 4 B-frags = 8 WMMA per K-step).
//   mode 0: Q * 0.125 -> (B=64, S=2048, F=64)
//   mode 1: K         -> (B, S, F)
//   mode 2: V         -> (B, F, S)   (transposed for PV B-fragments)
// ---------------------------------------------------------------------------
__global__ __launch_bounds__(128)
void mha_qkv_proj(const float* __restrict__ Xq, const float* __restrict__ Xk,
                  const float* __restrict__ Xv,
                  const bf16* __restrict__ WTq, const float* __restrict__ Bq,
                  const bf16* __restrict__ WTk, const float* __restrict__ Bk,
                  const bf16* __restrict__ WTv, const float* __restrict__ Bv,
                  bf16* __restrict__ Qws, bf16* __restrict__ Kws, bf16* __restrict__ Vtws)
{
  const int mode = blockIdx.z;
  const float* X  = (mode == 0) ? Xq  : (mode == 1) ? Xk  : Xv;
  const bf16*  WT = (mode == 0) ? WTq : (mode == 1) ? WTk : WTv;
  const float* Bi = (mode == 0) ? Bq  : (mode == 1) ? Bk  : Bv;

  const int n0   = blockIdx.x * 128;
  const int m0   = blockIdx.y * 64;
  const int tid  = threadIdx.x;
  const int wave = tid >> 5, lane = tid & 31;
  const int mq   = (wave & 1) * 32;         // wave's M offset (2 frags)
  const int nq   = (wave >> 1) * 64;        // wave's N offset (4 frags)

  __shared__ bf16 sA [2][ 64 * LDSTR];
  __shared__ bf16 sBT[2][128 * LDSTR];

  v8f acc00 = {}, acc01 = {}, acc02 = {}, acc03 = {};
  v8f acc10 = {}, acc11 = {}, acc12 = {}, acc13 = {};

  float4 ra[8];     // staged A tile (64x64 f32)
  v8bf   rb[8];     // staged B^T tile (128x64 bf16)

  auto load_regs = [&](int kt) {
    #pragma unroll
    for (int i = 0; i < 8; ++i) {
      int idx = tid + i * 128;              // 1024 float4 slots
      int row = idx >> 4, c4 = idx & 15;
      ra[i] = *(const float4*)(X + (size_t)(m0 + row) * 1024 + kt + c4 * 4);
    }
    #pragma unroll
    for (int i = 0; i < 8; ++i) {
      int idx = tid + i * 128;              // 1024 chunks of 8 bf16
      int n = idx >> 3, c8 = idx & 7;
      rb[i] = *(const v8bf*)(WT + (size_t)(n0 + n) * 1024 + kt + c8 * 8);
    }
  };
  auto store_lds = [&](int buf) {
    #pragma unroll
    for (int i = 0; i < 8; ++i) {
      int idx = tid + i * 128;
      int row = idx >> 4, c4 = idx & 15;
      v4bf o; o[0] = f2bf(ra[i].x); o[1] = f2bf(ra[i].y);
              o[2] = f2bf(ra[i].z); o[3] = f2bf(ra[i].w);
      *(v4bf*)&sA[buf][row * LDSTR + c4 * 4] = o;
    }
    #pragma unroll
    for (int i = 0; i < 8; ++i) {
      int idx = tid + i * 128;
      int n = idx >> 3, c8 = idx & 7;
      *(v8bf*)&sBT[buf][n * LDSTR + c8 * 8] = rb[i];
    }
  };

  load_regs(0);
  store_lds(0);
  for (int t = 0; t < 16; ++t) {
    const int cur = t & 1;
    if (t + 1 < 16) load_regs((t + 1) * 64);   // overlap with compute below
    __syncthreads();
    #pragma unroll
    for (int k0 = 0; k0 < 64; k0 += 32) {
      v16bf a0 = lds_load_a(&sA [cur][(mq +  0) * LDSTR + k0], LDSTR, lane);
      v16bf a1 = lds_load_a(&sA [cur][(mq + 16) * LDSTR + k0], LDSTR, lane);
      v16bf b0 = lds_load_b(&sBT[cur][(nq +  0) * LDSTR + k0], LDSTR, lane);
      v16bf b1 = lds_load_b(&sBT[cur][(nq + 16) * LDSTR + k0], LDSTR, lane);
      v16bf b2 = lds_load_b(&sBT[cur][(nq + 32) * LDSTR + k0], LDSTR, lane);
      v16bf b3 = lds_load_b(&sBT[cur][(nq + 48) * LDSTR + k0], LDSTR, lane);
      acc00 = WMMA_BF16(a0, b0, acc00);
      acc01 = WMMA_BF16(a0, b1, acc01);
      acc02 = WMMA_BF16(a0, b2, acc02);
      acc03 = WMMA_BF16(a0, b3, acc03);
      acc10 = WMMA_BF16(a1, b0, acc10);
      acc11 = WMMA_BF16(a1, b1, acc11);
      acc12 = WMMA_BF16(a1, b2, acc12);
      acc13 = WMMA_BF16(a1, b3, acc13);
    }
    if (t + 1 < 16) store_lds(cur ^ 1);
  }

  const int lo = lane & 15, hi = lane >> 4;
  auto store_frag = [&](v8f a, int moff, int noff) {
    int c = n0 + nq + noff + lo;              // output column 0..1023
    int h = c >> 6, f = c & 63;
    float bv = Bi[c];
    #pragma unroll
    for (int r = 0; r < 8; ++r) {
      int m  = m0 + mq + moff + r + hi * 8;   // token row 0..8191
      int s  = m >> 2, nb = m & 3;
      int bh = nb * 16 + h;                   // batch-head index 0..63
      float val = a[r] + bv;
      if (mode == 0)      Qws [((size_t)bh * 2048 + s) * 64 + f] = f2bf(val * 0.125f);
      else if (mode == 1) Kws [((size_t)bh * 2048 + s) * 64 + f] = f2bf(val);
      else                Vtws[((size_t)bh * 64 + f) * 2048 + s] = f2bf(val);
    }
  };
  store_frag(acc00,  0,  0); store_frag(acc01,  0, 16);
  store_frag(acc02,  0, 32); store_frag(acc03,  0, 48);
  store_frag(acc10, 16,  0); store_frag(acc11, 16, 16);
  store_frag(acc12, 16, 32); store_frag(acc13, 16, 48);
}

// ---------------------------------------------------------------------------
// Kernel 2: flash attention. K/V tiles move global->LDS via the Tensor Data
// Mover (double-buffered, one issuing wave, TENSORcnt sync) when available;
// falls back to register-staged copies otherwise.
// ---------------------------------------------------------------------------
__global__ __launch_bounds__(128)
void mha_attention(const bf16* __restrict__ Qws, const bf16* __restrict__ Kws,
                   const bf16* __restrict__ Vtws, bf16* __restrict__ AO)
{
  const int b   = blockIdx.y;         // batch-head 0..63
  const int m0  = blockIdx.x * 64;    // query row tile
  const int tid = threadIdx.x, wave = tid >> 5, lane = tid & 31;

  __shared__ bf16 sK[2][64 * LDSTR];  // rows: key idx, cols: F   (B^T for scores)
  __shared__ bf16 sV[2][64 * LDSTR];  // rows: F, cols: key idx   (B^T for PV)
  __shared__ bf16 sP[4][16 * LDSTR];  // per-wave P staging (16 x 64)

  // Q A-fragments for this wave's 16 rows, kept in registers (global stride 64)
  v16bf qa0, qa1;
  {
    const bf16* qbase = Qws + ((size_t)b * 2048 + m0 + wave * 16) * 64;
    int r = lane & 15, kb = (lane >> 4) << 3;
    const bf16* p = qbase + r * 64 + kb;
    union { v16bf v; v8bf h[2]; } u0, u1;
    u0.h[0] = *(const v8bf*)(p);        u0.h[1] = *(const v8bf*)(p + 16);
    u1.h[0] = *(const v8bf*)(p + 32);   u1.h[1] = *(const v8bf*)(p + 48);
    qa0 = u0.v; qa1 = u1.v;
  }

#if HAVE_TDM
  auto issue_tiles = [&](int jb, int buf) {   // wave-uniform; called by wave 0
    tdm_load_tile_bf16(Kws  + ((size_t)b * 2048 + jb) * 64,
                       (unsigned)(size_t)&sK[buf][0], 64, 64, 64);
    tdm_load_tile_bf16(Vtws + (size_t)b * 64 * 2048 + jb,
                       (unsigned)(size_t)&sV[buf][0], 64, 64, 2048);
  };
#else
  v8bf rk[4], rv[4];
  auto load_regs = [&](int jb) {
    #pragma unroll
    for (int i = 0; i < 4; ++i) {
      int idx = tid + i * 128;          // 512 chunks of 8 bf16
      int row = idx >> 3, c8 = idx & 7;
      rk[i] = *(const v8bf*)(Kws  + ((size_t)b * 2048 + jb + row) * 64 + c8 * 8);
      rv[i] = *(const v8bf*)(Vtws + ((size_t)b * 64 + row) * 2048 + jb + c8 * 8);
    }
  };
  auto store_lds = [&](int buf) {
    #pragma unroll
    for (int i = 0; i < 4; ++i) {
      int idx = tid + i * 128;
      int row = idx >> 3, c8 = idx & 7;
      *(v8bf*)&sK[buf][row * LDSTR + c8 * 8] = rk[i];
      *(v8bf*)&sV[buf][row * LDSTR + c8 * 8] = rv[i];
    }
  };
#endif

  float rowmax[8], rowsum[8];
  v8f o0 = {}, o1 = {}, o2 = {}, o3 = {};
  #pragma unroll
  for (int r = 0; r < 8; ++r) { rowmax[r] = -1e30f; rowsum[r] = 0.0f; }

#if HAVE_TDM
  if (wave == 0) issue_tiles(0, 0);
#else
  load_regs(0);
  store_lds(0);
#endif

  for (int j = 0; j < 32; ++j) {
    const int cur = j & 1;
#if HAVE_TDM
    if (wave == 0) __builtin_amdgcn_s_wait_tensorcnt(0);  // tile j resident
    __syncthreads();                   // all waves done with buffer cur^1 reads
    if (wave == 0 && j + 1 < 32) issue_tiles((j + 1) * 64, cur ^ 1);
#else
    if (j + 1 < 32) load_regs((j + 1) * 64);  // overlap with everything below
    __syncthreads();
#endif

    // scores S (16 x 64): Q already carries 1/sqrt(F)
    v8f s0 = {}, s1 = {}, s2 = {}, s3 = {};
    #define SCORE(sv, nb) { \
      v16bf b0_ = lds_load_b(&sK[cur][(nb) * LDSTR +  0], LDSTR, lane); \
      v16bf b1_ = lds_load_b(&sK[cur][(nb) * LDSTR + 32], LDSTR, lane); \
      sv = WMMA_BF16(qa0, b0_, sv); sv = WMMA_BF16(qa1, b1_, sv); }
    SCORE(s0,  0) SCORE(s1, 16) SCORE(s2, 32) SCORE(s3, 48)
    #undef SCORE

    // online softmax; C-layout rows live in 16-lane halves -> xor masks 1..8
    #pragma unroll
    for (int r = 0; r < 8; ++r) {
      float mx = fmaxf(fmaxf(s0[r], s1[r]), fmaxf(s2[r], s3[r]));
      #pragma unroll
      for (int d = 1; d < 16; d <<= 1) mx = fmaxf(mx, __shfl_xor(mx, d, 32));
      float nm   = fmaxf(rowmax[r], mx);
      float corr = __expf(rowmax[r] - nm);
      rowmax[r]  = nm;
      float p0 = __expf(s0[r] - nm), p1 = __expf(s1[r] - nm);
      float p2 = __expf(s2[r] - nm), p3 = __expf(s3[r] - nm);
      s0[r] = p0; s1[r] = p1; s2[r] = p2; s3[r] = p3;
      float ls = p0 + p1 + p2 + p3;
      #pragma unroll
      for (int d = 1; d < 16; d <<= 1) ls += __shfl_xor(ls, d, 32);
      rowsum[r] = rowsum[r] * corr + ls;
      o0[r] *= corr; o1[r] *= corr; o2[r] *= corr; o3[r] *= corr;
    }

    // stage P (C-layout f32 -> row-major bf16) into per-wave LDS region
    {
      int n = lane & 15, hi = lane >> 4;
      #pragma unroll
      for (int r = 0; r < 8; ++r) {
        int m = r + hi * 8;
        sP[wave][m * LDSTR +  0 + n] = f2bf(s0[r]);
        sP[wave][m * LDSTR + 16 + n] = f2bf(s1[r]);
        sP[wave][m * LDSTR + 32 + n] = f2bf(s2[r]);
        sP[wave][m * LDSTR + 48 + n] = f2bf(s3[r]);
      }
    }
    __syncthreads();

    v16bf pa0 = lds_load_a(&sP[wave][ 0], LDSTR, lane);
    v16bf pa1 = lds_load_a(&sP[wave][32], LDSTR, lane);
    #define PV(ov, nb) { \
      v16bf b0_ = lds_load_b(&sV[cur][(nb) * LDSTR +  0], LDSTR, lane); \
      v16bf b1_ = lds_load_b(&sV[cur][(nb) * LDSTR + 32], LDSTR, lane); \
      ov = WMMA_BF16(pa0, b0_, ov); ov = WMMA_BF16(pa1, b1_, ov); }
    PV(o0,  0) PV(o1, 16) PV(o2, 32) PV(o3, 48)
    #undef PV

#if !HAVE_TDM
    if (j + 1 < 32) store_lds(cur ^ 1);
#endif
  }

  // normalize and scatter to AO (token-major, column = head*64 + f)
  const int n_ = b >> 4;   // token batch index 0..3
  const int h_ = b & 15;   // head
  const int lo = lane & 15, hi = lane >> 4;
  auto wr = [&](v8f o, int noff) {
    int f = noff + lo;
    int c = h_ * 64 + f;
    #pragma unroll
    for (int r = 0; r < 8; ++r) {
      int s   = m0 + wave * 16 + r + hi * 8;
      int tok = s * 4 + n_;
      AO[(size_t)tok * 1024 + c] = f2bf(o[r] / rowsum[r]);
    }
  };
  wr(o0, 0); wr(o1, 16); wr(o2, 32); wr(o3, 48);
}

// ---------------------------------------------------------------------------
// Kernel 3: output projection, double-buffered, tile M64 x N128.
// AO(8192x1024 bf16) * Wo^T + bo -> f32
// ---------------------------------------------------------------------------
__global__ __launch_bounds__(128)
void mha_out_proj(const bf16* __restrict__ AO, const bf16* __restrict__ WOT,
                  const float* __restrict__ Bo, float* __restrict__ out)
{
  const int n0   = blockIdx.x * 128;
  const int m0   = blockIdx.y * 64;
  const int tid  = threadIdx.x, wave = tid >> 5, lane = tid & 31;
  const int mq   = (wave & 1) * 32;
  const int nq   = (wave >> 1) * 64;

  __shared__ bf16 sA [2][ 64 * LDSTR];
  __shared__ bf16 sBT[2][128 * LDSTR];

  v8f acc00 = {}, acc01 = {}, acc02 = {}, acc03 = {};
  v8f acc10 = {}, acc11 = {}, acc12 = {}, acc13 = {};

  v8bf ra[4], rb[8];
  auto load_regs = [&](int kt) {
    #pragma unroll
    for (int i = 0; i < 4; ++i) {
      int idx = tid + i * 128;
      int row = idx >> 3, c8 = idx & 7;
      ra[i] = *(const v8bf*)(AO + (size_t)(m0 + row) * 1024 + kt + c8 * 8);
    }
    #pragma unroll
    for (int i = 0; i < 8; ++i) {
      int idx = tid + i * 128;
      int n = idx >> 3, c8 = idx & 7;
      rb[i] = *(const v8bf*)(WOT + (size_t)(n0 + n) * 1024 + kt + c8 * 8);
    }
  };
  auto store_lds = [&](int buf) {
    #pragma unroll
    for (int i = 0; i < 4; ++i) {
      int idx = tid + i * 128;
      int row = idx >> 3, c8 = idx & 7;
      *(v8bf*)&sA[buf][row * LDSTR + c8 * 8] = ra[i];
    }
    #pragma unroll
    for (int i = 0; i < 8; ++i) {
      int idx = tid + i * 128;
      int n = idx >> 3, c8 = idx & 7;
      *(v8bf*)&sBT[buf][n * LDSTR + c8 * 8] = rb[i];
    }
  };

  load_regs(0);
  store_lds(0);
  for (int t = 0; t < 16; ++t) {
    const int cur = t & 1;
    if (t + 1 < 16) load_regs((t + 1) * 64);
    __syncthreads();
    #pragma unroll
    for (int k0 = 0; k0 < 64; k0 += 32) {
      v16bf a0 = lds_load_a(&sA [cur][(mq +  0) * LDSTR + k0], LDSTR, lane);
      v16bf a1 = lds_load_a(&sA [cur][(mq + 16) * LDSTR + k0], LDSTR, lane);
      v16bf b0 = lds_load_b(&sBT[cur][(nq +  0) * LDSTR + k0], LDSTR, lane);
      v16bf b1 = lds_load_b(&sBT[cur][(nq + 16) * LDSTR + k0], LDSTR, lane);
      v16bf b2 = lds_load_b(&sBT[cur][(nq + 32) * LDSTR + k0], LDSTR, lane);
      v16bf b3 = lds_load_b(&sBT[cur][(nq + 48) * LDSTR + k0], LDSTR, lane);
      acc00 = WMMA_BF16(a0, b0, acc00);
      acc01 = WMMA_BF16(a0, b1, acc01);
      acc02 = WMMA_BF16(a0, b2, acc02);
      acc03 = WMMA_BF16(a0, b3, acc03);
      acc10 = WMMA_BF16(a1, b0, acc10);
      acc11 = WMMA_BF16(a1, b1, acc11);
      acc12 = WMMA_BF16(a1, b2, acc12);
      acc13 = WMMA_BF16(a1, b3, acc13);
    }
    if (t + 1 < 16) store_lds(cur ^ 1);
  }

  const int lo = lane & 15, hi = lane >> 4;
  auto wr = [&](v8f a, int moff, int noff) {
    int c = n0 + nq + noff + lo;
    float bv = Bo[c];
    #pragma unroll
    for (int r = 0; r < 8; ++r) {
      int m = m0 + mq + moff + r + hi * 8;
      out[(size_t)m * 1024 + c] = a[r] + bv;
    }
  };
  wr(acc00,  0,  0); wr(acc01,  0, 16); wr(acc02,  0, 32); wr(acc03,  0, 48);
  wr(acc10, 16,  0); wr(acc11, 16, 16); wr(acc12, 16, 32); wr(acc13, 16, 48);
}

// ---------------------------------------------------------------------------
extern "C" void kernel_launch(void* const* d_in, const int* in_sizes, int n_in,
                              void* d_out, int out_size, void* d_ws, size_t ws_size,
                              hipStream_t stream) {
  (void)in_sizes; (void)n_in; (void)out_size; (void)ws_size;

  const float* query = (const float*)d_in[0];
  const float* key_t = (const float*)d_in[1];
  const float* value = (const float*)d_in[2];
  const float* wq_w  = (const float*)d_in[3];
  const float* wq_b  = (const float*)d_in[4];
  const float* wk_w  = (const float*)d_in[5];
  const float* wk_b  = (const float*)d_in[6];
  const float* wv_w  = (const float*)d_in[7];
  const float* wv_b  = (const float*)d_in[8];
  const float* wo_w  = (const float*)d_in[9];
  const float* wo_b  = (const float*)d_in[10];

  char* ws = (char*)d_ws;
  const size_t MB = 1024 * 1024;
  bf16* Qws  = (bf16*)(ws);              // 16 MB (64,2048,64)
  bf16* Kws  = (bf16*)(ws + 16 * MB);    // 16 MB (64,2048,64)
  bf16* Vtws = (bf16*)(ws + 32 * MB);    // 16 MB (64,64,2048)
  bf16* AO   = (bf16*)(ws + 48 * MB);    // 16 MB (8192,1024)
  bf16* WTq  = (bf16*)(ws + 64 * MB);    //  2 MB each, bf16 W^T
  bf16* WTk  = (bf16*)(ws + 66 * MB);
  bf16* WTv  = (bf16*)(ws + 68 * MB);
  bf16* WOT  = (bf16*)(ws + 70 * MB);

  dim3 gtr(16, 16, 4);
  mha_transpose_w<<<gtr, 256, 0, stream>>>(wq_w, wk_w, wv_w, wo_w,
                                           WTq, WTk, WTv, WOT);

  dim3 gproj(8, 128, 3);      // (N tiles of 128, M tiles of 64, Q/K/V)
  mha_qkv_proj<<<gproj, 128, 0, stream>>>(query, key_t, value,
                                          WTq, wq_b, WTk, wk_b, WTv, wv_b,
                                          Qws, Kws, Vtws);

  dim3 gattn(32, 64, 1);      // (query row tiles, batch-heads)
  mha_attention<<<gattn, 128, 0, stream>>>(Qws, Kws, Vtws, AO);

  dim3 gout(8, 128, 1);
  mha_out_proj<<<gout, 128, 0, stream>>>(AO, WOT, wo_b, (float*)d_out);
}
